// LSTM_64312840291090
// MI455X (gfx1250) — matrix operand order repaired and checked by reading, not compiled, not run on hardware
//
#include <hip/hip_runtime.h>

#define T_SZ   4096
#define IN_SZ  1024
#define H_SZ   1024
#define OUT_SZ 512
#define G4     (4 * H_SZ)
#define NBLK   128

typedef __attribute__((ext_vector_type(16))) __bf16 v16bf;
typedef __attribute__((ext_vector_type(8)))  float  v8f;

// ---------- helpers ----------
__device__ __forceinline__ unsigned short f2bf(float f) {
    unsigned u = __float_as_uint(f);
    u += 0x7fffu + ((u >> 16) & 1u);           // round-to-nearest-even
    return (unsigned short)(u >> 16);
}
__device__ __forceinline__ float bflo(unsigned u) { return __uint_as_float(u << 16); }
__device__ __forceinline__ float bfhi(unsigned u) { return __uint_as_float(u & 0xffff0000u); }

__device__ __forceinline__ void grid_barrier(unsigned* bar, unsigned nblk) {
    __threadfence();                            // publish this thread's stores
    __syncthreads();
    if (threadIdx.x == 0) {
        atomicAdd(bar, 1u);
        volatile unsigned* vb = bar;
        while (*vb < nblk) { __builtin_amdgcn_s_sleep(1); }
    }
    __syncthreads();
    __threadfence();                            // acquire other blocks' stores
}

// ---------- init / conversion kernels ----------
__global__ void init_state(unsigned* __restrict__ bars, unsigned short* __restrict__ hb, int nbars) {
    int i = blockIdx.x * blockDim.x + threadIdx.x;
    if (i < nbars) bars[i] = 0u;
    if (i < H_SZ)  hb[i]   = 0;
}

__global__ void cvt_x(const float* __restrict__ x, unsigned short* __restrict__ Xb) {
    int i = (blockIdx.x * blockDim.x + threadIdx.x) * 4;
    float4 v = *(const float4*)(x + i);
    unsigned short* o = Xb + i;
    o[0] = f2bf(v.x); o[1] = f2bf(v.y); o[2] = f2bf(v.z); o[3] = f2bf(v.w);
}

// rows packed [f; i; c; o] : Whb = cols [0,H), Wxb = cols [H, H+IN)
__global__ void cvt_gates(const float* __restrict__ wf, const float* __restrict__ wi,
                          const float* __restrict__ wc, const float* __restrict__ wo,
                          const float* __restrict__ bfv, const float* __restrict__ biv,
                          const float* __restrict__ bcv, const float* __restrict__ bov,
                          unsigned short* __restrict__ Whb, unsigned short* __restrict__ Wxb,
                          float* __restrict__ bias4) {
    int row = blockIdx.x;                       // 0..4095
    int g = row >> 10, j = row & (H_SZ - 1);
    const float* w = (g == 0) ? wf : (g == 1) ? wi : (g == 2) ? wc : wo;
    const float* b = (g == 0) ? bfv : (g == 1) ? biv : (g == 2) ? bcv : bov;
    const float* src = w + (size_t)j * (H_SZ + IN_SZ);
    for (int k = threadIdx.x; k < H_SZ + IN_SZ; k += blockDim.x) {
        unsigned short v = f2bf(src[k]);
        if (k < H_SZ) Whb[(size_t)row * H_SZ + k] = v;
        else          Wxb[(size_t)row * IN_SZ + (k - H_SZ)] = v;
    }
    if (threadIdx.x == 0) bias4[row] = b[j];
}

__global__ void cvt_why(const float* __restrict__ why, unsigned short* __restrict__ Wyb) {
    int row = blockIdx.x;
    const float* src = why + (size_t)row * H_SZ;
    unsigned short* dst = Wyb + (size_t)row * H_SZ;
    for (int k = threadIdx.x; k < H_SZ; k += blockDim.x) dst[k] = f2bf(src[k]);
}

// ---------- WMMA GEMM: GX(T x 4H) = Xb(T x IN) * Wxb^T ----------
__global__ void gx_gemm(const unsigned short* __restrict__ Xb,
                        const unsigned short* __restrict__ Wxb,
                        float* __restrict__ GX) {
    const int wave = (blockIdx.x * blockDim.x + threadIdx.x) >> 5;
    const int lane = threadIdx.x & 31;
    const int r = lane & 15, hi = lane >> 4;
    const int tilesN = G4 / 16;                 // 256
    const int tm = wave / tilesN, tn = wave % tilesN;
    const unsigned short* arow = Xb  + (size_t)(tm * 16 + r) * IN_SZ;
    const unsigned short* brow = Wxb + (size_t)(tn * 16 + r) * IN_SZ;
    v8f acc0 = {}, acc1 = {};
    for (int k0 = 0; k0 < IN_SZ; k0 += 64) {
        __builtin_prefetch(arow + k0 + 256, 0, 3);
        __builtin_prefetch(brow + k0 + 256, 0, 3);
        union { v16bf v; uint4 q[2]; } a, b;
        // A 16x32 bf16: low lanes K {0..7,16..23}, high lanes {8..15,24..31}
        a.q[0] = *(const uint4*)(arow + k0 + hi * 8);
        a.q[1] = *(const uint4*)(arow + k0 + 16 + hi * 8);
        // B 32x16 bf16: low lanes K {0..7,16..23}, high lanes {8..15,24..31}, col = n0 + r
        b.q[0] = *(const uint4*)(brow + k0 + hi * 8);
        b.q[1] = *(const uint4*)(brow + k0 + 16 + hi * 8);
        acc0 = __builtin_amdgcn_wmma_f32_16x16x32_bf16(false, a.v, false, b.v,
                                                       (short)0, acc0, false, false);
        a.q[0] = *(const uint4*)(arow + k0 + 32 + hi * 8);
        a.q[1] = *(const uint4*)(arow + k0 + 48 + hi * 8);
        b.q[0] = *(const uint4*)(brow + k0 + 32 + hi * 8);
        b.q[1] = *(const uint4*)(brow + k0 + 48 + hi * 8);
        acc1 = __builtin_amdgcn_wmma_f32_16x16x32_bf16(false, a.v, false, b.v,
                                                       (short)0, acc1, false, false);
    }
    float* crow = GX + (size_t)(tm * 16) * G4 + tn * 16 + r;
#pragma unroll
    for (int v = 0; v < 8; ++v) {
        int row = hi ? v + 8 : v;               // C layout: lanes 0-15 -> M=v, 16-31 -> M=v+8
        crow[(size_t)row * G4] = acc0[v] + acc1[v];
    }
}

// ---------- WMMA GEMM: Y(T x OUT) = Hb(T x H) * Wyb^T + by ----------
__global__ void y_gemm(const unsigned short* __restrict__ Hb,
                       const unsigned short* __restrict__ Wyb,
                       const float* __restrict__ by,
                       float* __restrict__ Y) {
    const int wave = (blockIdx.x * blockDim.x + threadIdx.x) >> 5;
    const int lane = threadIdx.x & 31;
    const int r = lane & 15, hi = lane >> 4;
    const int tilesN = OUT_SZ / 16;             // 32
    const int tm = wave / tilesN, tn = wave % tilesN;
    const unsigned short* arow = Hb  + (size_t)(tm * 16 + r) * H_SZ;
    const unsigned short* brow = Wyb + (size_t)(tn * 16 + r) * H_SZ;
    v8f acc0 = {}, acc1 = {};
    for (int k0 = 0; k0 < H_SZ; k0 += 64) {
        __builtin_prefetch(arow + k0 + 256, 0, 3);
        union { v16bf v; uint4 q[2]; } a, b;
        a.q[0] = *(const uint4*)(arow + k0 + hi * 8);
        a.q[1] = *(const uint4*)(arow + k0 + 16 + hi * 8);
        b.q[0] = *(const uint4*)(brow + k0 + hi * 8);
        b.q[1] = *(const uint4*)(brow + k0 + 16 + hi * 8);
        acc0 = __builtin_amdgcn_wmma_f32_16x16x32_bf16(false, a.v, false, b.v,
                                                       (short)0, acc0, false, false);
        a.q[0] = *(const uint4*)(arow + k0 + 32 + hi * 8);
        a.q[1] = *(const uint4*)(arow + k0 + 48 + hi * 8);
        b.q[0] = *(const uint4*)(brow + k0 + 32 + hi * 8);
        b.q[1] = *(const uint4*)(brow + k0 + 48 + hi * 8);
        acc1 = __builtin_amdgcn_wmma_f32_16x16x32_bf16(false, a.v, false, b.v,
                                                       (short)0, acc1, false, false);
    }
    int col = tn * 16 + r;
    float bias = by[col];
    float* cp = Y + (size_t)(tm * 16) * OUT_SZ + col;
#pragma unroll
    for (int v = 0; v < 8; ++v) {
        int row = hi ? v + 8 : v;
        cp[(size_t)row * OUT_SZ] = acc0[v] + acc1[v] + bias;
    }
}

// ---------- persistent recurrent kernel: one warp per hidden unit ----------
// 128 blocks x 256 threads = 1024 wave32 warps = H units.
// Warp j, lane group g = lane>>3 computes gate g of unit j (128 K-elems per lane).
// Lane 0 gathers the 4 gate pre-activations via shuffles, updates c (register),
// writes bf16 h[j]. Exactly ONE grid barrier per timestep.
__global__ void lstm_rec(const unsigned short* __restrict__ Whb,
                         const float* __restrict__ bias4,
                         const float* __restrict__ GX,
                         unsigned short* __restrict__ Hb,
                         unsigned short* __restrict__ hb,
                         unsigned* __restrict__ bars,
                         float* __restrict__ out_hc) {
    const int gid  = blockIdx.x * blockDim.x + threadIdx.x;   // 0..32767
    const int j    = gid >> 5;                                 // hidden unit, 0..1023
    const int lane = threadIdx.x & 31;
    const int g    = lane >> 3;                                // gate 0..3 (f,i,c,o)
    const int sub  = lane & 7;                                 // K-slice within gate
    const int row  = g * H_SZ + j;                             // packed gate row
    const unsigned short* wrow = Whb + (size_t)row * H_SZ + sub * 128;
    const float brow = bias4[row];
    float creg = 0.f;                                          // cell state (lane 0 only)

    for (int t = 0; t < T_SZ; ++t) {
        // dot: Wh[row, sub*128 .. +128) . h   (Wh bf16, L2-resident)
        float sum = 0.f;
        const unsigned short* hp = hb + sub * 128;
#pragma unroll 4
        for (int k = 0; k < 128; k += 8) {
            uint4 wq = *(const uint4*)(wrow + k);
            uint4 hq = *(const uint4*)(hp + k);
            sum = fmaf(bflo(wq.x), bflo(hq.x), sum);
            sum = fmaf(bfhi(wq.x), bfhi(hq.x), sum);
            sum = fmaf(bflo(wq.y), bflo(hq.y), sum);
            sum = fmaf(bfhi(wq.y), bfhi(hq.y), sum);
            sum = fmaf(bflo(wq.z), bflo(hq.z), sum);
            sum = fmaf(bfhi(wq.z), bfhi(hq.z), sum);
            sum = fmaf(bflo(wq.w), bflo(hq.w), sum);
            sum = fmaf(bfhi(wq.w), bfhi(hq.w), sum);
        }
        // reduce the 8 K-slices of each gate
        for (int off = 4; off; off >>= 1) sum += __shfl_down(sum, off, 8);
        float p = 0.f;
        if (sub == 0) p = sum + GX[(size_t)t * G4 + row] + brow;  // lanes 0,8,16,24

        // warp-local gate gather + cell/hidden update (no global round-trip)
        float pf = __shfl(p, 0, 32);
        float pi = __shfl(p, 8, 32);
        float pc = __shfl(p, 16, 32);
        float po = __shfl(p, 24, 32);
        if (lane == 0) {
            float fg = 1.f / (1.f + __expf(-pf));
            float ig = 1.f / (1.f + __expf(-pi));
            float gg = tanhf(pc);
            float og = 1.f / (1.f + __expf(-po));
            creg = fg * creg + ig * gg;
            float hn = og * tanhf(creg);
            unsigned short h16 = f2bf(hn);
            hb[j] = h16;
            Hb[(size_t)t * H_SZ + j] = h16;
            if (t == T_SZ - 1) { out_hc[j] = hn; out_hc[H_SZ + j] = creg; }
        }
        // publish h before anyone starts step t+1
        grid_barrier(&bars[t], NBLK);
    }
}

// ---------- host ----------
extern "C" void kernel_launch(void* const* d_in, const int* in_sizes, int n_in,
                              void* d_out, int out_size, void* d_ws, size_t ws_size,
                              hipStream_t stream) {
    const float* x   = (const float*)d_in[0];
    const float* wf  = (const float*)d_in[1];
    const float* bfv = (const float*)d_in[2];
    const float* wi  = (const float*)d_in[3];
    const float* biv = (const float*)d_in[4];
    const float* wc  = (const float*)d_in[5];
    const float* bcv = (const float*)d_in[6];
    const float* wo  = (const float*)d_in[7];
    const float* bov = (const float*)d_in[8];
    const float* why = (const float*)d_in[9];
    const float* byv = (const float*)d_in[10];
    float* out = (float*)d_out;

    char* ws = (char*)d_ws;
    size_t off = 0;
    auto take = [&](size_t bytes) {
        char* p = ws + off;
        off += (bytes + 255) & ~(size_t)255;
        return p;
    };
    unsigned short* Xb    = (unsigned short*)take((size_t)T_SZ * IN_SZ * 2);   // 8 MB
    unsigned short* Wxb   = (unsigned short*)take((size_t)G4 * IN_SZ * 2);     // 8 MB
    unsigned short* Whb   = (unsigned short*)take((size_t)G4 * H_SZ * 2);      // 8 MB
    unsigned short* Wyb   = (unsigned short*)take((size_t)OUT_SZ * H_SZ * 2);  // 1 MB
    float*          bias4 = (float*)take((size_t)G4 * 4);
    float*          GX    = (float*)take((size_t)T_SZ * G4 * 4);               // 64 MB
    unsigned short* Hb    = (unsigned short*)take((size_t)T_SZ * H_SZ * 2);    // 8 MB
    unsigned short* hb    = (unsigned short*)take((size_t)H_SZ * 2);
    unsigned*       bars  = (unsigned*)take((size_t)T_SZ * 4);

    init_state<<<32, 256, 0, stream>>>(bars, hb, T_SZ);
    cvt_x<<<(T_SZ * IN_SZ / 4) / 256, 256, 0, stream>>>(x, Xb);
    cvt_gates<<<G4, 256, 0, stream>>>(wf, wi, wc, wo, bfv, biv, bcv, bov, Whb, Wxb, bias4);
    cvt_why<<<OUT_SZ, 256, 0, stream>>>(why, Wyb);
    gx_gemm<<<(T_SZ / 16) * (G4 / 16) / 8, 256, 0, stream>>>(Xb, Wxb, GX);
    lstm_rec<<<NBLK, 256, 0, stream>>>(Whb, bias4, GX, Hb, hb, bars,
                                       out + (size_t)T_SZ * OUT_SZ);
    y_gemm<<<(T_SZ / 16) * (OUT_SZ / 16) / 8, 256, 0, stream>>>(Hb, Wyb, byv, out);
}
